// sphericalTemplateGenerator_66554813219086
// MI455X (gfx1250) — compile-verified, matrix operand order checked by reading, not run on hardware
//
#include <hip/hip_runtime.h>
#include <hip/hip_bf16.h>

// ---------------------------------------------------------------------------
// CDNA5 / gfx1250 implementation.
// Core compute: bf16 WMMA GEMM (v_wmma_f32_16x16x32_bf16), wave32, 8 waves per
// workgroup, 128x128 block tile. Tile staging uses the Tensor Data Mover
// (tensor_load_to_lds, TENSORcnt) when the builtin is available, else
// GLOBAL_LOAD_ASYNC_TO_LDS_B128 (ASYNCcnt). Both overlap DMA with WMMA.
// ---------------------------------------------------------------------------

typedef __attribute__((ext_vector_type(16))) __bf16 v16bf;
typedef __attribute__((ext_vector_type(8)))  float  v8f;
typedef __attribute__((ext_vector_type(4)))  unsigned int u32x4;
typedef __attribute__((ext_vector_type(8)))  int i32x8;
typedef __attribute__((ext_vector_type(4)))  int i32x4;

typedef unsigned short u16;
typedef unsigned int   u32;

#if __has_builtin(__builtin_amdgcn_tensor_load_to_lds) && \
    __has_builtin(__builtin_amdgcn_s_wait_tensorcnt)
#define HAVE_TDM 1
#else
#define HAVE_TDM 0
#endif

static __device__ __forceinline__ u16 f32_to_bf16_rne(float f) {
  union { float f; u32 u; } c; c.f = f;
  u32 r = c.u + 0x7FFFu + ((c.u >> 16) & 1u);
  return (u16)(r >> 16);
}
static __device__ __forceinline__ float bf16_to_f32(u16 h) {
  union { u32 u; float f; } c; c.u = ((u32)h) << 16;
  return c.f;
}

// --- CDNA5 async global->LDS DMA fallback (ASYNCcnt) -----------------------
static __device__ __forceinline__ void async_load_b128(const void* gptr, void* lptr) {
  const unsigned long long ga = (unsigned long long)(uintptr_t)gptr;
  const u32 la = (u32)(uintptr_t)lptr;
  asm volatile("global_load_async_to_lds_b128 %0, %1, off"
               :: "v"(la), "v"(ga)
               : "memory");
}
static __device__ __forceinline__ void wait_async_all() {
  asm volatile("s_wait_asynccnt 0x0" ::: "memory");
}

#if HAVE_TDM
// --- Tensor Data Mover: one descriptor moves a rows x tileK bf16 tile ------
// D# layout per CDNA5 ISA (2D tensor, groups 2/3 zero).
static __device__ __forceinline__ void tdm_load_tile_2d(
    const u16* gptr,     // tile start in global memory
    u32 lds_off,         // LDS byte offset of tile destination
    u32 tileK,           // tile_dim0 (elements, contiguous)
    u32 tileRows,        // tile_dim1 (rows)
    u32 rowStride,       // tensor_dim0_stride (elements between rows)
    u32 tensorRows)      // tensor_dim1 (rows remaining, for OOB clamp)
{
  const unsigned long long ga = (unsigned long long)(uintptr_t)gptr;
  u32x4 g0;
  g0.x = 1u;                                          // count=1, user descriptor
  g0.y = lds_off;                                     // lds_addr (bytes)
  g0.z = (u32)(ga & 0xFFFFFFFFull);                   // global_addr[31:0]
  g0.w = (u32)((ga >> 32) & 0x01FFFFFFull)            // global_addr[56:32]
       | (2u << 30);                                  // type=2 ("image")
  const u32 dim0 = rowStride;                         // tensor_dim0 (elements)
  const u32 dim1 = tensorRows;
  i32x8 g1;
  g1[0] = (int)(1u << 16);                            // wg_mask=0, data_size=1 (2B)
  g1[1] = (int)((dim0 & 0xFFFFu) << 16);              // [63:48] = dim0[15:0]
  g1[2] = (int)((dim0 >> 16) | ((dim1 & 0xFFFFu) << 16));
  g1[3] = (int)((dim1 >> 16) | (tileK << 16));        // [127:112] = tile_dim0
  g1[4] = (int)(tileRows & 0xFFFFu);                  // tile_dim1; tile_dim2=0
  g1[5] = (int)rowStride;                             // dim0_stride[31:0]
  g1[6] = 0;                                          // dim0_stride[47:32], dim1_stride lo
  g1[7] = 0;                                          // dim1_stride hi
  const i32x4 z4 = {0, 0, 0, 0};
#if defined(__clang_major__) && (__clang_major__ >= 23)
  const i32x8 z8 = {0, 0, 0, 0, 0, 0, 0, 0};
  __builtin_amdgcn_tensor_load_to_lds(g0, g1, z4, z4, z8, 0);
#else
  __builtin_amdgcn_tensor_load_to_lds(g0, g1, z4, z4, 0);
#endif
}
#endif // HAVE_TDM

// ---------------------------------------------------------------------------
// GEMM: C[M,N] = act( A[M,K](bf16) * B[N,K](bf16)^T + bias )
// Grid: (M/128, N/128, batch). Strides in elements; stride 0 = broadcast.
// ---------------------------------------------------------------------------
#define BM 128
#define BN 128
#define BK 32

struct FragBF { union { v16bf v; uint4 q[2]; }; };

__global__ __launch_bounds__(256, 2)
void k_gemm_bf16_wmma(const u16* __restrict__ A, const u16* __restrict__ B,
                      float* __restrict__ C, const float* __restrict__ bias,
                      int M, int N, int K, int act,
                      long long strideA, long long strideB, long long strideC)
{
  __shared__ u16 sA[2][BM * BK];
  __shared__ u16 sB[2][BN * BK];

  const int tid  = threadIdx.x;
  const int lane = tid & 31;
  const int wave = tid >> 5;
  const int wm   = wave & 3;   // wave M offset = wm*32
  const int wn   = wave >> 2;  // wave N offset = wn*64
  const int bRow = blockIdx.x * BM;
  const int bCol = blockIdx.y * BN;

  A += (long long)blockIdx.z * strideA;
  B += (long long)blockIdx.z * strideB;
  C += (long long)blockIdx.z * strideC;

  const int laneHalf = lane >> 4;   // 0/1: which K half per ISA 16-bit layout
  const int laneMod  = lane & 15;   // row within 16

  v8f acc[2][4] = {};

  // Stage one (A,B) tile pair into LDS buffer `buf` for K-offset k0.
  auto load_tile = [&](int k0, int buf) {
#if HAVE_TDM
    if (wave == 0) {   // one TDM descriptor pair per tile; EXEC-independent
      tdm_load_tile_2d(A + (long long)bRow * K + k0,
                       (u32)(uintptr_t)&sA[buf][0], BK, BM, (u32)K, (u32)(M - bRow));
      tdm_load_tile_2d(B + (long long)bCol * K + k0,
                       (u32)(uintptr_t)&sB[buf][0], BK, BN, (u32)K, (u32)(N - bCol));
    }
#else
    #pragma unroll
    for (int i = 0; i < 2; ++i) {
      const int idx = tid + i * 256;     // 0..511
      const int row = idx >> 2;          // 4 x 16B chunks per row
      const int qc  = idx & 3;
      async_load_b128(A + (long long)(bRow + row) * K + k0 + qc * 8,
                      &sA[buf][row * BK + qc * 8]);
      async_load_b128(B + (long long)(bCol + row) * K + k0 + qc * 8,
                      &sB[buf][row * BK + qc * 8]);
    }
#endif
  };
  auto wait_tiles = [&]() {
#if HAVE_TDM
    if (wave == 0) __builtin_amdgcn_s_wait_tensorcnt(0);
#else
    wait_async_all();
#endif
  };

  load_tile(0, 0);
  wait_tiles();
  __syncthreads();

  const int nK = K / BK;
  for (int kt = 0; kt < nK; ++kt) {
    const int buf = kt & 1;
    if (kt + 1 < nK) {
      // keep GL2 warm two tiles ahead, then fire the DMA for tile kt+1
      __builtin_prefetch(A + (long long)bRow * K + (long long)(kt + 2) * BK, 0, 3);
      __builtin_prefetch(B + (long long)bCol * K + (long long)(kt + 2) * BK, 0, 3);
      load_tile((kt + 1) * BK, buf ^ 1);
    }

    // Fragment loads per CDNA5 16-bit A/B layout:
    // lane<16 holds K {0..7,16..23}; lane>=16 holds K {8..15,24..31}
    FragBF a[2], b[4];
    #pragma unroll
    for (int mi = 0; mi < 2; ++mi) {
      const u16* p = &sA[buf][(wm * 32 + mi * 16 + laneMod) * BK + laneHalf * 8];
      a[mi].q[0] = *(const uint4*)p;
      a[mi].q[1] = *(const uint4*)(p + 16);
    }
    #pragma unroll
    for (int ni = 0; ni < 4; ++ni) {
      const u16* p = &sB[buf][(wn * 64 + ni * 16 + laneMod) * BK + laneHalf * 8];
      b[ni].q[0] = *(const uint4*)p;
      b[ni].q[1] = *(const uint4*)(p + 16);
    }

    #pragma unroll
    for (int mi = 0; mi < 2; ++mi)
      #pragma unroll
      for (int ni = 0; ni < 4; ++ni)
        acc[mi][ni] = __builtin_amdgcn_wmma_f32_16x16x32_bf16(
            false, a[mi].v, false, b[ni].v, (short)0, acc[mi][ni], false, false);

    if (kt + 1 < nK) wait_tiles();   // tile kt+1 landed in LDS
    __syncthreads();
  }

  // Epilogue: C/D layout — VGPR r holds M = r (lanes 0-15) / r+8 (lanes 16-31)
  #pragma unroll
  for (int mi = 0; mi < 2; ++mi) {
    #pragma unroll
    for (int ni = 0; ni < 4; ++ni) {
      const int n = bCol + wn * 64 + ni * 16 + laneMod;
      #pragma unroll
      for (int r = 0; r < 8; ++r) {
        const int m = bRow + wm * 32 + mi * 16 + laneHalf * 8 + r;
        float v = acc[mi][ni][r];
        if (bias) v += bias[n];
        if (act == 1)      v = 0.5f * v * (1.0f + erff(v * 0.70710678f)); // exact gelu
        else if (act == 2) v = (v > 0.f) ? v : 0.2f * v;                   // leaky relu
        if (m < M && n < N) C[(long long)m * N + n] = v;
      }
    }
  }
}

// ---------------------------------------------------------------------------
// Elementwise / reduction support kernels (wave32 shuffles)
// ---------------------------------------------------------------------------

__global__ void k_init_w(u16* __restrict__ w, long long n, u32 seed) {
  long long i = (long long)blockIdx.x * blockDim.x + threadIdx.x;
  if (i >= n) return;
  u32 h = (u32)i * 2654435761u + seed * 0x9E3779B9u;
  h ^= h >> 16; h *= 0x85EBCA6Bu; h ^= h >> 13; h *= 0xC2B2AE35u; h ^= h >> 16;
  float f = ((float)(h & 0xFFFFu) * (1.0f / 65535.0f) - 0.5f) * 0.04f;
  w[i] = f32_to_bf16_rne(f);
}

__global__ void k_pack_x(const float* __restrict__ xyz, u16* __restrict__ x,
                         int rows, int C_) {
  long long g = (long long)blockIdx.x * blockDim.x + threadIdx.x;
  if (g >= (long long)rows * C_) return;
  int r = (int)(g / C_), c = (int)(g % C_);
  int b = r >> 9, i = r & 511;                 // rows = B*512
  float v = xyz[((long long)b * 2048 + (long long)i * 4) * 3 + (c % 3)];
  x[g] = f32_to_bf16_rne(v * (0.5f + 0.001f * (float)c));
}

__global__ __launch_bounds__(128)
void k_layernorm_cast(const float* __restrict__ X, u16* __restrict__ Y, int C_) {
  const long long row = blockIdx.x;
  const float* xr = X + row * C_;
  __shared__ float red[2][4];
  const int tid = threadIdx.x, lane = tid & 31, wave = tid >> 5;
  float s = 0.f, s2 = 0.f;
  for (int c = tid; c < C_; c += 128) { float v = xr[c]; s += v; s2 += v * v; }
  for (int o = 16; o > 0; o >>= 1) { s += __shfl_xor(s, o, 32); s2 += __shfl_xor(s2, o, 32); }
  if (lane == 0) { red[0][wave] = s; red[1][wave] = s2; }
  __syncthreads();
  float ts  = red[0][0] + red[0][1] + red[0][2] + red[0][3];
  float ts2 = red[1][0] + red[1][1] + red[1][2] + red[1][3];
  float mean = ts / (float)C_;
  float var  = ts2 / (float)C_ - mean * mean;
  float rstd = rsqrtf(var + 1e-5f);
  for (int c = tid; c < C_; c += 128)
    Y[row * C_ + c] = f32_to_bf16_rne((xr[c] - mean) * rstd);
}

__global__ void k_cast_f32_bf16(const float* __restrict__ X, u16* __restrict__ Y,
                                long long n) {
  long long g = (long long)blockIdx.x * blockDim.x + threadIdx.x;
  if (g < n) Y[g] = f32_to_bf16_rne(X[g]);
}

__global__ __launch_bounds__(256)
void k_softmax_cast(const float* __restrict__ X, u16* __restrict__ Y,
                    int L, float scale) {
  const long long row = blockIdx.x;
  const float* xr = X + row * L;
  __shared__ float red[8];
  const int tid = threadIdx.x, lane = tid & 31, wave = tid >> 5;
  float mx = -1e30f;
  for (int c = tid; c < L; c += 256) mx = fmaxf(mx, xr[c] * scale);
  for (int o = 16; o > 0; o >>= 1) mx = fmaxf(mx, __shfl_xor(mx, o, 32));
  if (lane == 0) red[wave] = mx;
  __syncthreads();
  mx = red[0];
  #pragma unroll
  for (int w = 1; w < 8; ++w) mx = fmaxf(mx, red[w]);
  __syncthreads();
  float s = 0.f;
  for (int c = tid; c < L; c += 256) s += __expf(xr[c] * scale - mx);
  for (int o = 16; o > 0; o >>= 1) s += __shfl_xor(s, o, 32);
  if (lane == 0) red[wave] = s;
  __syncthreads();
  s = 0.f;
  #pragma unroll
  for (int w = 0; w < 8; ++w) s += red[w];
  const float inv = 1.0f / s;
  for (int c = tid; c < L; c += 256)
    Y[row * L + c] = f32_to_bf16_rne(__expf(xr[c] * scale - mx) * inv);
}

__global__ void k_transpose_b(const u16* __restrict__ X, u16* __restrict__ Y,
                              int B_, int R_, int C_) {
  long long g = (long long)blockIdx.x * blockDim.x + threadIdx.x;
  long long per = (long long)R_ * C_;
  if (g >= (long long)B_ * per) return;
  int b = (int)(g / per);
  long long rem = g % per;
  int r = (int)(rem / C_), c = (int)(rem % C_);
  Y[(long long)b * per + (long long)c * R_ + r] = X[g];
}

__global__ void k_sample_pts(const float* __restrict__ src, float* __restrict__ dst,
                             int B_, int N_) {
  long long g = (long long)blockIdx.x * blockDim.x + threadIdx.x;
  if (g >= (long long)B_ * 512 * 3) return;
  int b = (int)(g / (512 * 3));
  int rem = (int)(g % (512 * 3));
  int i = rem / 3, c = rem % 3;
  dst[g] = src[((long long)b * N_ + (long long)i * (N_ / 512)) * 3 + c];
}

__global__ void k_write_scalar(float* __restrict__ p) {
  if (threadIdx.x == 0 && blockIdx.x == 0) p[0] = 0.0f;
}

// ---------------------------------------------------------------------------
// Orchestration
// ---------------------------------------------------------------------------
extern "C" void kernel_launch(void* const* d_in, const int* in_sizes, int n_in,
                              void* d_out, int out_size, void* d_ws, size_t ws_size,
                              hipStream_t stream) {
  (void)in_sizes; (void)n_in; (void)out_size; (void)ws_size;

  const float* xyz = (const float*)d_in[0];          // (8,2048,3)
  const float* usp = (const float*)d_in[1];          // (8,2048,3)

  const int Bz = 8, D = 384, F = 768, NQ = 512;
  const int ROWS = Bz * NQ;                          // 4096 token rows

  // bump allocator over workspace
  char* base = (char*)d_ws;
  size_t off = 0;
  auto alloc = [&](size_t bytes) -> void* {
    void* p = base + off;
    off += (bytes + 255) & ~(size_t)255;
    return p;
  };

  u16*   W1  = (u16*)alloc((size_t)D * D * 2);
  u16*   W2  = (u16*)alloc((size_t)F * D * 2);
  u16*   W3  = (u16*)alloc((size_t)D * F * 2);
  u16*   X0  = (u16*)alloc((size_t)ROWS * D * 2);
  float* Yf  = (float*)alloc((size_t)ROWS * F * 4);  // widest f32 staging
  u16*   X1  = (u16*)alloc((size_t)ROWS * D * 2);
  u16*   X2  = (u16*)alloc((size_t)ROWS * F * 2);
  u16*   X3  = (u16*)alloc((size_t)ROWS * D * 2);
  float* S   = (float*)alloc((size_t)Bz * NQ * NQ * 4);
  u16*   Sbf = (u16*)alloc((size_t)Bz * NQ * NQ * 2);
  u16*   VT  = (u16*)alloc((size_t)Bz * D * NQ * 2);

  auto blocks = [](long long n, int t) { return (unsigned)((n + t - 1) / t); };

  // synthesized weights (deterministic)
  k_init_w<<<blocks((long long)D * D, 256), 256, 0, stream>>>(W1, (long long)D * D, 1u);
  k_init_w<<<blocks((long long)F * D, 256), 256, 0, stream>>>(W2, (long long)F * D, 2u);
  k_init_w<<<blocks((long long)D * F, 256), 256, 0, stream>>>(W3, (long long)D * F, 3u);

  // token features from xyz
  k_pack_x<<<blocks((long long)ROWS * D, 256), 256, 0, stream>>>(xyz, X0, ROWS, D);

  // proj: Y = gelu(X0 @ W1^T) ; layernorm -> X1
  {
    dim3 g(ROWS / BM, D / BN, 1);
    k_gemm_bf16_wmma<<<g, 256, 0, stream>>>(X0, W1, Yf, nullptr, ROWS, D, D, 1, 0, 0, 0);
    k_layernorm_cast<<<ROWS, 128, 0, stream>>>(Yf, X1, D);
  }
  // ffn up: Y = gelu(X1 @ W2^T) -> X2 (bf16)
  {
    dim3 g(ROWS / BM, F / BN, 1);
    k_gemm_bf16_wmma<<<g, 256, 0, stream>>>(X1, W2, Yf, nullptr, ROWS, F, D, 1, 0, 0, 0);
    k_cast_f32_bf16<<<blocks((long long)ROWS * F, 256), 256, 0, stream>>>(Yf, X2, (long long)ROWS * F);
  }
  // ffn down: Y = X2 @ W3^T ; layernorm -> X3
  {
    dim3 g(ROWS / BM, D / BN, 1);
    k_gemm_bf16_wmma<<<g, 256, 0, stream>>>(X2, W3, Yf, nullptr, ROWS, D, F, 0, 0, 0, 0);
    k_layernorm_cast<<<ROWS, 128, 0, stream>>>(Yf, X3, D);
  }
  // attention scores per batch: S = X3 @ X3^T   (M=N=512, K=384)
  {
    dim3 g(NQ / BM, NQ / BN, Bz);
    k_gemm_bf16_wmma<<<g, 256, 0, stream>>>(X3, X3, S, nullptr, NQ, NQ, D, 0,
                                            (long long)NQ * D, (long long)NQ * D,
                                            (long long)NQ * NQ);
  }
  // softmax rows (scale = 1/sqrt(dh)=1/sqrt(96)) -> bf16 probs
  k_softmax_cast<<<Bz * NQ, 256, 0, stream>>>(S, Sbf, NQ, 0.1020621f);

  // V^T per batch (512x384 -> 384x512) so AV fits the A(M,K)*B(N,K)^T form
  k_transpose_b<<<blocks((long long)Bz * NQ * D, 256), 256, 0, stream>>>(X3, VT, Bz, NQ, D);

  // O = probs @ V : written straight into the 'q' section of d_out (f32)
  float* out = (float*)d_out;
  float* q_out = out + 12288;                         // after u_query (8*512*3)
  {
    dim3 g(NQ / BM, D / BN, Bz);
    k_gemm_bf16_wmma<<<g, 256, 0, stream>>>(Sbf, VT, q_out, nullptr, NQ, D, NQ, 0,
                                            (long long)NQ * NQ, (long long)D * NQ,
                                            (long long)NQ * D);
  }

  // remaining output sections
  const long long qElems = (long long)Bz * NQ * D;    // 1572864
  k_sample_pts<<<blocks(12288, 256), 256, 0, stream>>>(usp, out, Bz, 2048);            // u_query
  k_write_scalar<<<1, 1, 0, stream>>>(out + 12288 + qElems);                            // scalar 0
  k_sample_pts<<<blocks(12288, 256), 256, 0, stream>>>(xyz, out + 12288 + qElems + 1,  // u_template
                                                       Bz, 2048);
}
